// SketchDepthColorizer_71073118814672
// MI455X (gfx1250) — matrix operand order repaired and checked by reading, not compiled, not run on hardware
//
#include <hip/hip_runtime.h>
#include <hip/hip_bf16.h>

typedef _Float16 f16;
typedef __attribute__((ext_vector_type(16))) _Float16 v16h;
typedef __attribute__((ext_vector_type(8)))  float    v8f;

#define USE_ASYNC_LDS 1

// ---------------------------------------------------------------------------
// Async global->LDS copy of one 16-byte chunk (CDNA5 GLOBAL_LOAD_ASYNC_TO_LDS,
// tracked by ASYNCcnt).  LDS address = ptrtoint of AS(3) pointer.
// ---------------------------------------------------------------------------
__device__ __forceinline__ unsigned lds_off_of(const void* p) {
  return (unsigned)(unsigned long long)(__attribute__((address_space(3))) const void*)p;
}

__device__ __forceinline__ void copy16_g2lds(f16* dstLds, const f16* srcG) {
#if USE_ASYNC_LDS
  asm volatile("global_load_async_to_lds_b128 %0, %1, off"
               :: "v"(lds_off_of(dstLds)), "v"((const void*)srcG)
               : "memory");
#else
  *(int4*)dstLds = *(const int4*)srcG;
#endif
}

__device__ __forceinline__ void async_wait_all() {
#if USE_ASYNC_LDS
#if __has_builtin(__builtin_amdgcn_s_wait_asynccnt)
  __builtin_amdgcn_s_wait_asynccnt(0);
#else
  asm volatile("s_wait_asynccnt 0x0" ::: "memory");
#endif
#endif
}

// ---------------------------------------------------------------------------
// CDNA5 WMMA 16x16x32 f16 VGPR layouts used here (wave32; A/B k order is contiguous, not ISA table order):
// A fragment: lane holds row M=lane&15; element e -> K = a_kidx(e, lane>>4)
//   (one contiguous 16-element K run: {16h..16h+15}).
// B fragment: lane holds col N=lane&15; element e -> K = (lane>>4)*16 + e.
// C/D: element j of lane l -> row = j + 8*(l>=16), col = l&15.
// ---------------------------------------------------------------------------
__device__ __forceinline__ int a_kidx(int e, int h) {
  return e + 16 * h;
}

// ---------------------------------------------------------------------------
// Weight packing: OIHW f32 -> fragment-ordered f16 panel [Kpad/32][OCpad][32]
// so a lane's 16 B-elements are 32B-contiguous in LDS after a raw slab copy.
// k = ic*kh*kw + ky*kw + kx
// ---------------------------------------------------------------------------
__global__ void pack_conv_w(const float* __restrict__ w, f16* __restrict__ pk,
                            int OC, int IC, int kh, int kw, int OCpad, int Kpad) {
  int idx = blockIdx.x * blockDim.x + threadIdx.x;
  int total = Kpad * OCpad;
  if (idx >= total) return;
  int kr = idx & 31;
  int t2 = idx >> 5;
  int oc = t2 % OCpad, kb = t2 / OCpad;
  int k = kb * 32 + kr;
  int khw = kh * kw, Kreal = IC * khw;
  float v = 0.f;
  if (k < Kreal && oc < OC) {
    int ic = k / khw, rem = k % khw, ky = rem / kw, kx = rem % kw;
    v = w[((oc * IC + ic) * kh + ky) * kw + kx];
  }
  pk[idx] = (f16)v;
}

// ConvTranspose2d(k=4,s=2,p=1) parity decomposition: output parity (py,px),
// taps ty,tx in {0,1}:  ky = 2*ty + 1 - py,  input offset dy = py - ty.
// weights are [IC][OC][4][4];  k = ic*4 + ty*2 + tx
__global__ void pack_deconv_w(const float* __restrict__ w, f16* __restrict__ pk,
                              int OC, int IC, int OCpad, int Kpad, int py, int px) {
  int idx = blockIdx.x * blockDim.x + threadIdx.x;
  int total = Kpad * OCpad;
  if (idx >= total) return;
  int kr = idx & 31;
  int t2 = idx >> 5;
  int oc = t2 % OCpad, kb = t2 / OCpad;
  int k = kb * 32 + kr;
  int Kreal = IC * 4;
  float v = 0.f;
  if (k < Kreal && oc < OC) {
    int ic = k >> 2, tap = k & 3, ty = tap >> 1, tx = tap & 1;
    int ky = 2 * ty + 1 - py;
    int kx = 2 * tx + 1 - px;
    v = w[((ic * OC + oc) * 4 + ky) * 4 + kx];
  }
  pk[idx] = (f16)v;
}

// ---------------------------------------------------------------------------
// Implicit-GEMM conv.  Block = 4 waves; wave w owns M tile (16 spatial rows)
// (blockIdx.x*4+w)*16 and accumulates ALL output-channel tiles (NT x 16) in
// registers.  A fragment is im2col-gathered DIRECTLY into registers (a 16x32
// A tile has zero intra-wave duplication), overlapping the async global->LDS
// DMA of the fragment-ordered B slab.  KH/KW/NT compile-time; spatial extents
// pow2 (shift/mask).
// ---------------------------------------------------------------------------
template <int KH, int KW, int NT>
__global__ __launch_bounds__(128)
void conv2d_wmma(const float* __restrict__ inA, const float* __restrict__ inB,
                 int cA, int cB, int bcast,
                 const f16* __restrict__ wpack, const float* __restrict__ bias,
                 float* __restrict__ out,
                 int Bn, int IH, int IW, int lOH, int lOW,
                 int OC, int Kreal, int Kpad, int stride, int pad, int relu) {
  constexpr int KHW = KH * KW;
  constexpr int OCpad = NT * 16;
  __shared__ __align__(32) f16 Bs[OCpad * 32];
  const int tid = threadIdx.x;
  const int wave = tid >> 5, lane = tid & 31, h = lane >> 4;
  const int OW = 1 << lOW, OH = 1 << lOH;
  const int M = Bn << (lOH + lOW);
  const int m0 = (blockIdx.x * 4 + wave) * 16;
  const int m = m0 + (lane & 15);
  const bool mok = m < M;
  const int ox = m & (OW - 1);
  const int oy = (m >> lOW) & (OH - 1);
  const int bi = m >> (lOW + lOH);
  v8f acc[NT];
#pragma unroll
  for (int i = 0; i < NT; ++i) acc[i] = {};
  for (int k0 = 0; k0 < Kpad; k0 += 32) {
    // B slab: contiguous fragment-ordered f16 -> async DMA into LDS
    for (int c = tid; c < 4 * OCpad; c += 128)
      copy16_g2lds(&Bs[c * 8], wpack + (size_t)(k0 >> 5) * (OCpad * 32) + c * 8);
    // A fragment: direct im2col gather (overlaps the async copy)
    v16h a;
#pragma unroll
    for (int e = 0; e < 16; ++e) {
      int k = k0 + a_kidx(e, h);
      float val = 0.f;
      if (mok && k < Kreal) {
        int ic = k / KHW, rem = k % KHW, ky = rem / KW, kx = rem % KW;
        int iy = oy * stride + ky - pad;
        int ix = ox * stride + kx - pad;
        if (iy >= 0 && iy < IH && ix >= 0 && ix < IW) {
          if (ic < cA)
            val = inA[(((long)bi * cA + ic) * IH + iy) * IW + ix];
          else if (bcast)
            val = inB[(long)bi * cB + (ic - cA)];
          else
            val = inB[(((long)bi * cB + (ic - cA)) * IH + iy) * IW + ix];
        }
      }
      a[e] = (f16)val;
    }
    async_wait_all();
    __syncthreads();
#pragma unroll
    for (int nt = 0; nt < NT; ++nt) {
      v16h b = *(const v16h*)(Bs + (nt * 16 + (lane & 15)) * 32 + h * 16);
      acc[nt] = __builtin_amdgcn_wmma_f32_16x16x32_f16(false, a, false, b,
                                                       (short)0, acc[nt], false, false);
    }
    __syncthreads();
  }
#pragma unroll
  for (int nt = 0; nt < NT; ++nt) {
    int n = nt * 16 + (lane & 15);
    if (n < OC) {
#pragma unroll
      for (int j = 0; j < 8; ++j) {
        int mj = m0 + j + 8 * h;
        if (mj < M) {
          int oxj = mj & (OW - 1);
          int oyj = (mj >> lOW) & (OH - 1);
          int bj = mj >> (lOW + lOH);
          float val = acc[nt][j] + bias[n];
          if (relu) val = fmaxf(val, 0.f);
          out[(((long)bj * OC + n) << (lOH + lOW)) + ((long)oyj << lOW) + oxj] = val;
        }
      }
    }
  }
}

// ---------------------------------------------------------------------------
// Transposed conv (k4 s2 p1), one parity class per launch.  M = Bn*IH*IW
// (u,v grid of the input); output at (2u+py, 2v+px).  K = IC*4.
// ---------------------------------------------------------------------------
template <int NT>
__global__ __launch_bounds__(128)
void deconv2d_wmma(const float* __restrict__ inA, const float* __restrict__ inB,
                   int cA, int cB,
                   const f16* __restrict__ wpack, const float* __restrict__ bias,
                   float* __restrict__ out,
                   int Bn, int lIH, int lIW, int OC, int Kreal, int Kpad,
                   int py, int px, int relu) {
  constexpr int OCpad = NT * 16;
  __shared__ __align__(32) f16 Bs[OCpad * 32];
  const int tid = threadIdx.x;
  const int wave = tid >> 5, lane = tid & 31, h = lane >> 4;
  const int IH = 1 << lIH, IW = 1 << lIW;
  const int M = Bn << (lIH + lIW);
  const int m0 = (blockIdx.x * 4 + wave) * 16;
  const int m = m0 + (lane & 15);
  const bool mok = m < M;
  const int vv = m & (IW - 1);
  const int u = (m >> lIW) & (IH - 1);
  const int bi = m >> (lIW + lIH);
  v8f acc[NT];
#pragma unroll
  for (int i = 0; i < NT; ++i) acc[i] = {};
  for (int k0 = 0; k0 < Kpad; k0 += 32) {
    for (int c = tid; c < 4 * OCpad; c += 128)
      copy16_g2lds(&Bs[c * 8], wpack + (size_t)(k0 >> 5) * (OCpad * 32) + c * 8);
    v16h a;
#pragma unroll
    for (int e = 0; e < 16; ++e) {
      int k = k0 + a_kidx(e, h);
      float val = 0.f;
      if (mok && k < Kreal) {
        int ic = k >> 2, tap = k & 3, ty = tap >> 1, tx = tap & 1;
        int iy = u + py - ty;
        int ix = vv + px - tx;
        if (iy >= 0 && iy < IH && ix >= 0 && ix < IW) {
          if (ic < cA)
            val = inA[(((long)bi * cA + ic) * IH + iy) * IW + ix];
          else
            val = inB[(((long)bi * cB + (ic - cA)) * IH + iy) * IW + ix];
        }
      }
      a[e] = (f16)val;
    }
    async_wait_all();
    __syncthreads();
#pragma unroll
    for (int nt = 0; nt < NT; ++nt) {
      v16h b = *(const v16h*)(Bs + (nt * 16 + (lane & 15)) * 32 + h * 16);
      acc[nt] = __builtin_amdgcn_wmma_f32_16x16x32_f16(false, a, false, b,
                                                       (short)0, acc[nt], false, false);
    }
    __syncthreads();
  }
#pragma unroll
  for (int nt = 0; nt < NT; ++nt) {
    int n = nt * 16 + (lane & 15);
    if (n < OC) {
#pragma unroll
      for (int j = 0; j < 8; ++j) {
        int mj = m0 + j + 8 * h;
        if (mj < M) {
          int vj = mj & (IW - 1);
          int uj = (mj >> lIW) & (IH - 1);
          int bj = mj >> (lIW + lIH);
          int oyj = 2 * uj + py, oxj = 2 * vj + px;
          float val = acc[nt][j] + bias[n];
          if (relu) val = fmaxf(val, 0.f);
          out[(((long)bj * OC + n) * (IH << 1) + oyj) * (IW << 1) + oxj] = val;
        }
      }
    }
  }
}

// ---------------------------------------------------------------------------
// Attention scores (per batch): S[n][m] = sum_c q[c][n]*k[c][m], C=8 pad 32.
// Both fragments gathered directly into registers -> no LDS, no barriers.
// ---------------------------------------------------------------------------
__global__ __launch_bounds__(128)
void attn_scores_wmma(const float* __restrict__ q, const float* __restrict__ kmat,
                      float* __restrict__ S, int N, int C) {
  const int tid = threadIdx.x;
  const int wave = tid >> 5, lane = tid & 31, h = lane >> 4;
  const int n0 = blockIdx.x * 16;
  const int m0 = blockIdx.y * 64 + wave * 16;
  const int nrow = n0 + (lane & 15);
  const int mcol = m0 + (lane & 15);
  v16h a, b;
#pragma unroll
  for (int e = 0; e < 16; ++e) {
    int ca = a_kidx(e, h);
    a[e] = (ca < C) ? (f16)q[(long)ca * N + nrow] : (f16)0.f;
    int cb = h * 16 + e;
    b[e] = (cb < C) ? (f16)kmat[(long)cb * N + mcol] : (f16)0.f;
  }
  v8f acc = {};
  acc = __builtin_amdgcn_wmma_f32_16x16x32_f16(false, a, false, b,
                                               (short)0, acc, false, false);
#pragma unroll
  for (int j = 0; j < 8; ++j)
    S[(long)(n0 + j + 8 * h) * N + mcol] = acc[j];
}

__global__ void softmax_rows(float* __restrict__ S, int N) {
  __shared__ float red[256];
  const int tid = threadIdx.x;
  float* p = S + (long)blockIdx.x * N;
  float mx = -1e30f;
  for (int i = tid; i < N; i += 256) mx = fmaxf(mx, p[i]);
  red[tid] = mx; __syncthreads();
  for (int s = 128; s > 0; s >>= 1) {
    if (tid < s) red[tid] = fmaxf(red[tid], red[tid + s]);
    __syncthreads();
  }
  mx = red[0]; __syncthreads();
  float sum = 0.f;
  for (int i = tid; i < N; i += 256) { float e = __expf(p[i] - mx); p[i] = e; sum += e; }
  red[tid] = sum; __syncthreads();
  for (int s = 128; s > 0; s >>= 1) {
    if (tid < s) red[tid] += red[tid + s];
    __syncthreads();
  }
  float inv = 1.f / red[0];
  for (int i = tid; i < N; i += 256) p[i] *= inv;
}

// o[c][n] = sum_m P[n][m] * v[c][m].  Block = 4 waves (4 n-tiles) sharing the
// column-major 64x32 V panel in LDS; each wave keeps all 64 channels in
// registers, so the 67 MB probability matrix is streamed exactly once, with
// P fragments gathered directly (one contiguous 16-float K run per lane).
__global__ __launch_bounds__(128)
void attn_out_wmma(const float* __restrict__ P, const float* __restrict__ vmat,
                   float* __restrict__ o, int N) {
  __shared__ __align__(32) f16 Bs[64 * 32];   // [c][kk]
  const int tid = threadIdx.x;
  const int wave = tid >> 5, lane = tid & 31, h = lane >> 4;
  const int n0 = (blockIdx.x * 4 + wave) * 16;
  const int nrow = n0 + (lane & 15);
  const int vc = tid >> 1, vk0 = (tid & 1) * 16;  // V-panel fill assignment
  v8f acc[4];
#pragma unroll
  for (int i = 0; i < 4; ++i) acc[i] = {};
  for (int m0 = 0; m0 < N; m0 += 32) {
    // V panel (f32 -> f16), vectorized: 16 contiguous halfs per thread
    v16h vt;
#pragma unroll
    for (int e = 0; e < 16; ++e) vt[e] = (f16)vmat[(long)vc * N + m0 + vk0 + e];
    *(v16h*)(Bs + vc * 32 + vk0) = vt;
    // P fragment: direct gather (contiguous runs)
    v16h a;
#pragma unroll
    for (int e = 0; e < 16; ++e)
      a[e] = (f16)P[(long)nrow * N + m0 + a_kidx(e, h)];
    __syncthreads();
#pragma unroll
    for (int nt = 0; nt < 4; ++nt) {
      v16h b = *(const v16h*)(Bs + (nt * 16 + (lane & 15)) * 32 + h * 16);
      acc[nt] = __builtin_amdgcn_wmma_f32_16x16x32_f16(false, a, false, b,
                                                       (short)0, acc[nt], false, false);
    }
    __syncthreads();
  }
#pragma unroll
  for (int nt = 0; nt < 4; ++nt) {
    int c = nt * 16 + (lane & 15);
#pragma unroll
    for (int j = 0; j < 8; ++j)
      o[(long)c * N + n0 + j + 8 * h] = acc[nt][j];
  }
}

// ---------------------------------------------------------------------------
// Small VALU kernels
// ---------------------------------------------------------------------------
__global__ void avgpool_to8(const float* __restrict__ x, float* __restrict__ y,
                            int Bn, int C, int H, int W) {
  int idx = blockIdx.x * blockDim.x + threadIdx.x;
  int total = Bn * C * 64;
  if (idx >= total) return;
  int px = idx & 7, t = idx >> 3, py = t & 7; t >>= 3;
  int c = t % C, b = t / C;
  int bh = H / 8, bw = W / 8;
  float s = 0.f;
  for (int yy = 0; yy < bh; ++yy)
    for (int xx = 0; xx < bw; ++xx)
      s += x[(((long)b * C + c) * H + py * bh + yy) * W + px * bw + xx];
  y[idx] = s / (float)(bh * bw);
}

__global__ void linear_relu(const float* __restrict__ x, const float* __restrict__ w,
                            const float* __restrict__ bias, float* __restrict__ y,
                            int Bn, int K, int O) {
  int idx = blockIdx.x * blockDim.x + threadIdx.x;
  if (idx >= Bn * O) return;
  int b = idx / O, o = idx % O;
  float s = bias[o];
  for (int i = 0; i < K; ++i) s += x[(long)b * K + i] * w[(long)o * K + i];
  y[idx] = fmaxf(s, 0.f);
}

// per-row (W-axis) unbiased std -> sigmoid mask
__global__ void row_mask(const float* __restrict__ x, float* __restrict__ mask,
                         int rows, int W) {
  int r = blockIdx.x * blockDim.x + threadIdx.x;
  if (r >= rows) return;
  const float* p = x + (long)r * W;
  float s = 0.f, s2 = 0.f;
  for (int i = 0; i < W; ++i) { float v = p[i]; s += v; s2 += v * v; }
  float mean = s / (float)W;
  float var = fmaxf(0.f, (s2 - (float)W * mean * mean) / (float)(W - 1));
  float sd = sqrtf(var);
  mask[r] = 1.f / (1.f + __expf(-(0.05f - sd) * 10.f));
}

// 5-tap vertical box blend with replicate padding; optional tanh epilogue
__global__ void line_apply(const float* __restrict__ x, const float* __restrict__ mask,
                           float* __restrict__ out, int BC, int H, int W,
                           float strength, int do_tanh) {
  long idx = (long)blockIdx.x * blockDim.x + threadIdx.x;
  long total = (long)BC * H * W;
  if (idx >= total) return;
  int w = idx % W;
  int h = (idx / W) % H;
  int bc = idx / ((long)W * H);
  float vs = 0.f, mm = 0.f;
  for (int d = -2; d <= 2; ++d) {
    int hh = h + d; hh = hh < 0 ? 0 : (hh >= H ? H - 1 : hh);
    vs += x[((long)bc * H + hh) * W + w];
    mm += mask[(long)bc * H + hh];
  }
  vs *= 0.2f;
  mm = mm * 0.2f * strength;
  float v = x[idx] * (1.f - mm) + vs * mm;
  if (do_tanh) v = tanhf(v);
  out[idx] = v;
}

__global__ void blend_inplace(float* __restrict__ a, const float* __restrict__ b,
                              float wa, float wb, long n) {
  long i = (long)blockIdx.x * blockDim.x + threadIdx.x;
  if (i < n) a[i] = a[i] * wa + b[i] * wb;
}

__global__ void attn_residual(float* __restrict__ e, const float* __restrict__ o,
                              const float* __restrict__ gamma, long n) {
  long i = (long)blockIdx.x * blockDim.x + threadIdx.x;
  if (i < n) e[i] = gamma[0] * o[i] + e[i];
}

// ---------------------------------------------------------------------------
extern "C" void kernel_launch(void* const* d_in, const int* in_sizes, int n_in,
                              void* d_out, int out_size, void* d_ws, size_t ws_size,
                              hipStream_t stream) {
  (void)in_sizes; (void)n_in; (void)out_size; (void)ws_size;
  const float* sketch = (const float*)d_in[0];
  const float* depth  = (const float*)d_in[1];
  const float* style  = (const float*)d_in[2];
  const float* enc1_w = (const float*)d_in[3];  const float* enc1_b = (const float*)d_in[4];
  const float* enc2_w = (const float*)d_in[5];  const float* enc2_b = (const float*)d_in[6];
  const float* enc3_w = (const float*)d_in[7];  const float* enc3_b = (const float*)d_in[8];
  const float* se1_w  = (const float*)d_in[9];  const float* se1_b  = (const float*)d_in[10];
  const float* se2_w  = (const float*)d_in[11]; const float* se2_b  = (const float*)d_in[12];
  const float* se3_w  = (const float*)d_in[13]; const float* se3_b  = (const float*)d_in[14];
  const float* sp1_w  = (const float*)d_in[15]; const float* sp1_b  = (const float*)d_in[16];
  const float* sp2_w  = (const float*)d_in[17]; const float* sp2_b  = (const float*)d_in[18];
  const float* sm_w   = (const float*)d_in[19]; const float* sm_b   = (const float*)d_in[20];
  const float* aq_w   = (const float*)d_in[21]; const float* aq_b   = (const float*)d_in[22];
  const float* ak_w   = (const float*)d_in[23]; const float* ak_b   = (const float*)d_in[24];
  const float* av_w   = (const float*)d_in[25]; const float* av_b   = (const float*)d_in[26];
  const float* agamma = (const float*)d_in[27];
  const float* bn_w   = (const float*)d_in[28]; const float* bn_b   = (const float*)d_in[29];
  const float* dec3_w = (const float*)d_in[30]; const float* dec3_b = (const float*)d_in[31];
  const float* dec2_w = (const float*)d_in[32]; const float* dec2_b = (const float*)d_in[33];
  const float* dec1_w = (const float*)d_in[34]; const float* dec1_b = (const float*)d_in[35];
  float* outp = (float*)d_out;

  const int B = 4, N = 4096;  // attention sequence length 64*64

  char* base = (char*)d_ws;
  size_t off = 0;
  auto alloc = [&](size_t bytes) -> void* {
    off = (off + 255) & ~(size_t)255;
    void* p = base + off; off += bytes; return p;
  };
  auto fal = [&](size_t elems) -> float* { return (float*)alloc(elems * 4); };
  auto hal = [&](size_t elems) -> f16*   { return (f16*)alloc(elems * 2); };
  auto lg2 = [](int v) { int l = 0; while ((1 << l) < v) ++l; return l; };

  // activations
  float* e1    = fal((size_t)B * 16 * 256 * 256);
  float* e2    = fal((size_t)B * 32 * 128 * 128);
  float* e3    = fal((size_t)B * 64 * 64 * 64);
  float* e3f   = fal((size_t)B * 64 * 64 * 64);
  float* s1    = fal((size_t)B * 16 * 256 * 256);
  float* s2    = fal((size_t)B * 32 * 128 * 128);
  float* s3    = fal((size_t)B * 64 * 64 * 64);
  float* pooled= fal((size_t)B * 64 * 64);
  float* sf1   = fal((size_t)B * 64);
  float* sf2   = fal((size_t)B * 64);
  float* e3ws  = fal((size_t)B * 64 * 64 * 64);
  float* qb    = fal((size_t)B * 8 * N);
  float* kb    = fal((size_t)B * 8 * N);
  float* vb    = fal((size_t)B * 64 * N);
  float* Sbuf  = fal((size_t)N * N);          // one batch at a time (67 MB, L2-resident)
  float* ao    = fal((size_t)B * 64 * N);
  float* bfeat = fal((size_t)B * 64 * 64 * 64);
  float* d3    = fal((size_t)B * 32 * 128 * 128);
  float* d2    = fal((size_t)B * 16 * 256 * 256);
  float* d2f   = fal((size_t)B * 16 * 256 * 256);
  float* preo  = fal((size_t)B * 3 * 256 * 256);
  float* maskb = fal((size_t)16384);          // reused per line-filter
  // packed f16 weight panels (fragment order)
  f16* pk_enc1 = hal(32 * 16);
  f16* pk_enc2 = hal(256 * 32);
  f16* pk_enc3 = hal(512 * 64);
  f16* pk_se1  = hal(32 * 16);
  f16* pk_se2  = hal(256 * 32);
  f16* pk_se3  = hal(512 * 64);
  f16* pk_sm   = hal(1152 * 64);
  f16* pk_q    = hal(64 * 16);
  f16* pk_k    = hal(64 * 16);
  f16* pk_v    = hal(64 * 64);
  f16* pk_bn   = hal(576 * 64);
  f16* pk_d3[4]; for (int i = 0; i < 4; ++i) pk_d3[i] = hal(512 * 32);
  f16* pk_d2[4]; for (int i = 0; i < 4; ++i) pk_d2[i] = hal(256 * 16);
  f16* pk_dec1 = hal(288 * 16);

  auto packc = [&](const float* w, f16* pk, int OC, int IC, int kh, int kw,
                   int OCpad, int Kpad) {
    int total = Kpad * OCpad;
    pack_conv_w<<<(total + 255) / 256, 256, 0, stream>>>(w, pk, OC, IC, kh, kw, OCpad, Kpad);
  };
  auto packd = [&](const float* w, f16* pk, int OC, int IC, int OCpad, int Kpad,
                   int py, int px) {
    int total = Kpad * OCpad;
    pack_deconv_w<<<(total + 255) / 256, 256, 0, stream>>>(w, pk, OC, IC, OCpad, Kpad, py, px);
  };
  auto conv = [&](const float* inA, const float* inB, int cA, int cB, int bc,
                  const f16* pk, const float* bias, float* op,
                  int IH, int IW, int OH, int OW, int OC, int OCpad,
                  int Kreal, int Kpad, int kh, int kw, int st, int pad, int relu) {
    int M = B * OH * OW;
    dim3 g((M + 63) / 64), blk(128);
    int lOH = lg2(OH), lOW = lg2(OW);
#define CARGS inA, inB, cA, cB, bc, pk, bias, op, B, IH, IW, lOH, lOW, OC, Kreal, Kpad, st, pad, relu
    if (kh == 3 && OCpad == 16)       conv2d_wmma<3, 3, 1><<<g, blk, 0, stream>>>(CARGS);
    else if (kh == 3)                 conv2d_wmma<3, 3, 4><<<g, blk, 0, stream>>>(CARGS);
    else if (kh == 4 && OCpad == 32)  conv2d_wmma<4, 4, 2><<<g, blk, 0, stream>>>(CARGS);
    else if (kh == 4)                 conv2d_wmma<4, 4, 4><<<g, blk, 0, stream>>>(CARGS);
    else if (OCpad == 16)             conv2d_wmma<1, 1, 1><<<g, blk, 0, stream>>>(CARGS);
    else                              conv2d_wmma<1, 1, 4><<<g, blk, 0, stream>>>(CARGS);
#undef CARGS
  };
  auto deconv = [&](const float* inA, const float* inB, int cA, int cB,
                    f16* const* pk, const float* bias, float* op,
                    int IH, int IW, int OC, int OCpad, int Kreal, int Kpad) {
    int M = B * IH * IW;
    dim3 g((M + 63) / 64), blk(128);
    int lIH = lg2(IH), lIW = lg2(IW);
    for (int py = 0; py < 2; ++py)
      for (int px = 0; px < 2; ++px) {
        if (OCpad == 16)
          deconv2d_wmma<1><<<g, blk, 0, stream>>>(inA, inB, cA, cB, pk[py * 2 + px],
                                                  bias, op, B, lIH, lIW, OC, Kreal, Kpad, py, px, 1);
        else
          deconv2d_wmma<2><<<g, blk, 0, stream>>>(inA, inB, cA, cB, pk[py * 2 + px],
                                                  bias, op, B, lIH, lIW, OC, Kreal, Kpad, py, px, 1);
      }
  };
  auto linef = [&](const float* x, float* op, int C, int H, int W,
                   float strength, int do_tanh) {
    int rows = B * C * H;
    row_mask<<<(rows + 255) / 256, 256, 0, stream>>>(x, maskb, rows, W);
    long total = (long)B * C * H * W;
    line_apply<<<(unsigned)((total + 255) / 256), 256, 0, stream>>>(
        x, maskb, op, B * C, H, W, strength, do_tanh);
  };

  // ---- pack all weights ----
  packc(enc1_w, pk_enc1, 16, 2, 3, 3, 16, 32);
  packc(enc2_w, pk_enc2, 32, 16, 4, 4, 32, 256);
  packc(enc3_w, pk_enc3, 64, 32, 4, 4, 64, 512);
  packc(se1_w,  pk_se1, 16, 3, 3, 3, 16, 32);
  packc(se2_w,  pk_se2, 32, 16, 4, 4, 32, 256);
  packc(se3_w,  pk_se3, 64, 32, 4, 4, 64, 512);
  packc(sm_w,   pk_sm, 64, 128, 3, 3, 64, 1152);
  packc(aq_w,   pk_q, 8, 64, 1, 1, 16, 64);
  packc(ak_w,   pk_k, 8, 64, 1, 1, 16, 64);
  packc(av_w,   pk_v, 64, 64, 1, 1, 64, 64);
  packc(bn_w,   pk_bn, 64, 64, 3, 3, 64, 576);
  for (int py = 0; py < 2; ++py)
    for (int px = 0; px < 2; ++px) {
      packd(dec3_w, pk_d3[py * 2 + px], 32, 128, 32, 512, py, px);
      packd(dec2_w, pk_d2[py * 2 + px], 16, 64, 16, 256, py, px);
    }
  packc(dec1_w, pk_dec1, 3, 32, 3, 3, 16, 288);

  // ---- encoder ----
  conv(sketch, depth, 1, 1, 0, pk_enc1, enc1_b, e1, 256, 256, 256, 256, 16, 16, 18, 32, 3, 3, 1, 1, 1);
  conv(e1, e1, 16, 0, 0, pk_enc2, enc2_b, e2, 256, 256, 128, 128, 32, 32, 256, 256, 4, 4, 2, 1, 1);
  conv(e2, e2, 32, 0, 0, pk_enc3, enc3_b, e3, 128, 128, 64, 64, 64, 64, 512, 512, 4, 4, 2, 1, 1);
  linef(e3, e3f, 64, 64, 64, 0.8f, 0);

  // ---- style branch ----
  conv(style, style, 3, 0, 0, pk_se1, se1_b, s1, 256, 256, 256, 256, 16, 16, 27, 32, 3, 3, 1, 1, 1);
  conv(s1, s1, 16, 0, 0, pk_se2, se2_b, s2, 256, 256, 128, 128, 32, 32, 256, 256, 4, 4, 2, 1, 1);
  conv(s2, s2, 32, 0, 0, pk_se3, se3_b, s3, 128, 128, 64, 64, 64, 64, 512, 512, 4, 4, 2, 1, 1);
  avgpool_to8<<<(B * 64 * 64 + 255) / 256, 256, 0, stream>>>(s3, pooled, B, 64, 64, 64);
  linear_relu<<<1, 256, 0, stream>>>(pooled, sp1_w, sp1_b, sf1, B, 4096, 64);
  linear_relu<<<1, 256, 0, stream>>>(sf1, sp2_w, sp2_b, sf2, B, 64, 64);
  // style mix: concat(e3f, broadcast(sf2)) -> 3x3 conv, then 0.99/0.01 blend
  conv(e3f, sf2, 64, 64, 1, pk_sm, sm_b, e3ws, 64, 64, 64, 64, 64, 64, 1152, 1152, 3, 3, 1, 1, 1);
  {
    long n = (long)B * 64 * N;
    blend_inplace<<<(unsigned)((n + 255) / 256), 256, 0, stream>>>(e3f, e3ws, 0.99f, 0.01f, n);
  }

  // ---- self-attention (batch-sequential to keep S in one 67 MB buffer) ----
  conv(e3f, e3f, 64, 0, 0, pk_q, aq_b, qb, 64, 64, 64, 64, 8, 16, 64, 64, 1, 1, 1, 0, 0);
  conv(e3f, e3f, 64, 0, 0, pk_k, ak_b, kb, 64, 64, 64, 64, 8, 16, 64, 64, 1, 1, 1, 0, 0);
  conv(e3f, e3f, 64, 0, 0, pk_v, av_b, vb, 64, 64, 64, 64, 64, 64, 64, 64, 1, 1, 1, 0, 0);
  for (int b = 0; b < B; ++b) {
    const float* qp = qb + (long)b * 8 * N;
    const float* kp = kb + (long)b * 8 * N;
    const float* vp = vb + (long)b * 64 * N;
    float* op = ao + (long)b * 64 * N;
    attn_scores_wmma<<<dim3(N / 16, N / 64), 128, 0, stream>>>(qp, kp, Sbuf, N, 8);
    softmax_rows<<<N, 256, 0, stream>>>(Sbuf, N);
    attn_out_wmma<<<dim3(N / 64), 128, 0, stream>>>(Sbuf, vp, op, N);
  }
  {
    long n = (long)B * 64 * N;
    attn_residual<<<(unsigned)((n + 255) / 256), 256, 0, stream>>>(e3f, ao, agamma, n);
  }

  // ---- decoder ----
  conv(e3f, e3f, 64, 0, 0, pk_bn, bn_b, bfeat, 64, 64, 64, 64, 64, 64, 576, 576, 3, 3, 1, 1, 1);
  deconv(bfeat, e3f, 64, 64, pk_d3, dec3_b, d3, 64, 64, 32, 32, 512, 512);
  deconv(d3, e2, 32, 32, pk_d2, dec2_b, d2, 128, 128, 16, 16, 256, 256);
  linef(d2, d2f, 16, 256, 256, 0.8f, 0);
  conv(d2f, e1, 16, 16, 0, pk_dec1, dec1_b, preo, 256, 256, 256, 256, 3, 16, 288, 288, 3, 3, 1, 1, 0);
  linef(preo, outp, 3, 256, 256, 0.9f, 1);
}
